// QuadrupletLoss_80161269612715
// MI455X (gfx1250) — compile-verified, matrix operand order checked by reading, not run on hardware
//
#include <hip/hip_runtime.h>
#include <hip/hip_bf16.h>
#include <cstdint>
#include <cstddef>

#define NN 2048
#define DD 2048
#define K_HARD 10
#define MARGIN1 0.3f
#define MARGIN2 0.3f
#define CENTER_W 0.01f
#define HALF_CNT 2097152u  // (NN*NN)/2 for threefry counter split

typedef __bf16 bf16_t;
typedef bf16_t v8bf  __attribute__((ext_vector_type(8)));
typedef bf16_t v16bf __attribute__((ext_vector_type(16)));
typedef float  v8f   __attribute__((ext_vector_type(8)));
typedef unsigned int u32x4 __attribute__((ext_vector_type(4)));
typedef int          i32x8 __attribute__((ext_vector_type(8)));
typedef int          i32x4 __attribute__((ext_vector_type(4)));

// ---------- helpers ----------
static __device__ __forceinline__ uint16_t f32_to_bf16_rne(float f) {
  uint32_t u = __builtin_bit_cast(uint32_t, f);
  uint32_t r = u + 0x7fffu + ((u >> 16) & 1u);
  return (uint16_t)(r >> 16);
}

static __device__ __forceinline__ void threefry2x32(uint32_t x0, uint32_t x1,
                                                    uint32_t& o0, uint32_t& o1) {
  // jax.random.key(42) -> keypair (0, 42)
  const uint32_t k0 = 0u, k1 = 42u;
  const uint32_t ksv[3] = {k0, k1, k0 ^ k1 ^ 0x1BD11BDAu};
  x0 += ksv[0]; x1 += ksv[1];
  const int rot[2][4] = {{13, 15, 26, 6}, {17, 29, 16, 24}};
#pragma unroll
  for (int i = 0; i < 5; ++i) {
#pragma unroll
    for (int j = 0; j < 4; ++j) {
      int r = rot[i & 1][j];
      x0 += x1;
      x1 = (x1 << r) | (x1 >> (32 - r));
      x1 ^= x0;
    }
    x0 += ksv[(i + 1) % 3];
    x1 += ksv[(i + 2) % 3] + (uint32_t)(i + 1);
  }
  o0 = x0; o1 = x1;
}

static __device__ __forceinline__ float jax_uniform01(uint32_t bits) {
  float f = __builtin_bit_cast(float, (bits >> 9) | 0x3f800000u) - 1.0f;
  return fmaxf(0.0f, f);
}

static __device__ __forceinline__ bool finitef(float x) {
  return fabsf(x) < 3.0e38f;
}

// Issue one 2-D TDM tile load (bf16 elements) into LDS.
// D# layout per CDNA5 ISA ch.8: group0 = {count/mode, lds_addr, global_addr, type},
// group1 = {data_size, tensor dims, tile dims, dim0 stride}; groups 2/3 zero (2-D).
// This toolchain exposes the 6-arg builtin (extra trailing int32x8 group, zero-filled).
static __device__ __forceinline__ void tdm_load_tile_bf16(
    uint32_t lds_byte_off, const uint16_t* gsrc,
    uint32_t tile_h, uint32_t row_stride_elems) {
  uint64_t ga = (uint64_t)(uintptr_t)gsrc;
  u32x4 g0;
  g0.x = 1u;                                                  // count=1, user mode
  g0.y = lds_byte_off;                                        // LDS dest (bytes)
  g0.z = (uint32_t)ga;                                        // global addr [31:0]
  g0.w = (uint32_t)((ga >> 32) & 0x1FFFFFFu) | (2u << 30);    // addr[56:32] | type=2
  i32x8 g1;
  g1[0] = (int)(1u << 16);                                    // wg_mask=0, data_size=1 (2B)
  g1[1] = (int)((DD & 0xFFFFu) << 16);                        // tensor_dim0[15:0]
  g1[2] = (int)(((DD >> 16) & 0xFFFFu) | ((NN & 0xFFFFu) << 16));   // dim0 hi | dim1 lo
  g1[3] = (int)(((NN >> 16) & 0xFFFFu) | (32u << 16));        // dim1 hi | tile_dim0=32
  g1[4] = (int)(tile_h & 0xFFFFu);                            // tile_dim1 | tile_dim2=0
  g1[5] = (int)row_stride_elems;                              // tensor_dim0_stride[31:0]
  g1[6] = 0;                                                  // stride0 hi | dim1_stride lo
  g1[7] = 0;
  i32x4 gz4 = {0, 0, 0, 0};
  i32x8 gz8 = {0, 0, 0, 0, 0, 0, 0, 0};
  __builtin_amdgcn_tensor_load_to_lds(g0, g1, gz4, gz4, gz8, 0);
}

// ---------- kernel 0: zero accumulators ----------
__global__ void init_kernel(float* accum) {
  if (threadIdx.x < 4) accum[threadIdx.x] = 0.0f;
}

// ---------- kernel 1: bf16 hi/lo split + row squared norms ----------
__global__ __launch_bounds__(256)
void prep_kernel(const float* __restrict__ X, uint16_t* __restrict__ Xhi,
                 uint16_t* __restrict__ Xlo, float* __restrict__ sq) {
  __shared__ float red[256];
  const int row = blockIdx.x;
  const float* xr = X + (size_t)row * DD;
  float acc = 0.f;
  for (int j = threadIdx.x; j < DD; j += 256) {
    float v = xr[j];
    uint16_t h = f32_to_bf16_rne(v);
    float hf = __builtin_bit_cast(float, (uint32_t)h << 16);
    uint16_t l = f32_to_bf16_rne(v - hf);
    Xhi[(size_t)row * DD + j] = h;
    Xlo[(size_t)row * DD + j] = l;
    acc += v * v;
  }
  red[threadIdx.x] = acc;
  __syncthreads();
  for (int s = 128; s > 0; s >>= 1) {
    if ((int)threadIdx.x < s) red[threadIdx.x] += red[threadIdx.x + s];
    __syncthreads();
  }
  if (threadIdx.x == 0) sq[row] = red[0];
}

// ---------- kernel 2: Gram via TDM-staged LDS tiles + bf16x3 WMMA ----------
// Block tile 128(rows) x 64(cols); 8 waves, each computes 32x32.
// Wave 0 drives the Tensor Data Mover; LDS double-buffered.
__global__ __launch_bounds__(256)
void gram_dist_kernel(const uint16_t* __restrict__ Xhi, const uint16_t* __restrict__ Xlo,
                      const float* __restrict__ sq, float* __restrict__ dist) {
  __shared__ __align__(16) uint16_t sAhi[2][128 * 32];
  __shared__ __align__(16) uint16_t sAlo[2][128 * 32];
  __shared__ __align__(16) uint16_t sBhi[2][64 * 32];
  __shared__ __align__(16) uint16_t sBlo[2][64 * 32];

  const int lane = threadIdx.x & 31;
  const int wave = threadIdx.x >> 5;
  const int waveRow = wave & 3;   // 0..3
  const int waveCol = wave >> 2;  // 0..1
  const int rowBlock = blockIdx.y * 128;
  const int colBlock = blockIdx.x * 64;

  const int lmod  = lane & 15;
  const int khalf = (lane >> 4) * 8;   // 0 or 8, per 16-bit A-matrix layout

  v8f acc[2][2] = {};

  // prologue: DMA first K-slab into buffer 0
  if (wave == 0) {
    size_t aOff = (size_t)rowBlock * DD;     // + kb (=0)
    size_t bOff = (size_t)colBlock * DD;
    tdm_load_tile_bf16((uint32_t)(uintptr_t)&sAhi[0][0], Xhi + aOff, 128, DD);
    tdm_load_tile_bf16((uint32_t)(uintptr_t)&sAlo[0][0], Xlo + aOff, 128, DD);
    tdm_load_tile_bf16((uint32_t)(uintptr_t)&sBhi[0][0], Xhi + bOff, 64, DD);
    tdm_load_tile_bf16((uint32_t)(uintptr_t)&sBlo[0][0], Xlo + bOff, 64, DD);
  }

  for (int kb = 0; kb < DD; kb += 32) {
    const int cur = (kb >> 5) & 1;
    if (wave == 0) {
      if (kb + 32 < DD) {
        const int nxt = cur ^ 1;
        size_t aOff = (size_t)rowBlock * DD + (size_t)(kb + 32);
        size_t bOff = (size_t)colBlock * DD + (size_t)(kb + 32);
        tdm_load_tile_bf16((uint32_t)(uintptr_t)&sAhi[nxt][0], Xhi + aOff, 128, DD);
        tdm_load_tile_bf16((uint32_t)(uintptr_t)&sAlo[nxt][0], Xlo + aOff, 128, DD);
        tdm_load_tile_bf16((uint32_t)(uintptr_t)&sBhi[nxt][0], Xhi + bOff, 64, DD);
        tdm_load_tile_bf16((uint32_t)(uintptr_t)&sBlo[nxt][0], Xlo + bOff, 64, DD);
        // 8 outstanding (4 cur + 4 nxt); in-order completion => <=4 means cur done
        __builtin_amdgcn_s_wait_tensorcnt(4);
      } else {
        __builtin_amdgcn_s_wait_tensorcnt(0);
      }
    }
    __syncthreads();  // LDS tiles for `cur` visible to all waves

    v16bf ah[2], al[2], bh[2], bl[2];
#pragma unroll
    for (int t = 0; t < 2; ++t) {
      // A fragment: LDS tile row-major [row][32]; per-lane K = khalf..+7 and +16..+23
      int arow = waveRow * 32 + t * 16 + lmod;
      const uint16_t* ap = &sAhi[cur][arow * 32 + khalf];
      v8bf a0 = *reinterpret_cast<const v8bf*>(ap);
      v8bf a1 = *reinterpret_cast<const v8bf*>(ap + 16);
      ah[t] = __builtin_shufflevector(a0, a1, 0,1,2,3,4,5,6,7,8,9,10,11,12,13,14,15);
      const uint16_t* cp = &sAlo[cur][arow * 32 + khalf];
      v8bf c0 = *reinterpret_cast<const v8bf*>(cp);
      v8bf c1 = *reinterpret_cast<const v8bf*>(cp + 16);
      al[t] = __builtin_shufflevector(c0, c1, 0,1,2,3,4,5,6,7,8,9,10,11,12,13,14,15);
      // B fragment: B = X^T tile -> lane n holds row (colBlock+n) of X, same pattern
      int brow = waveCol * 32 + t * 16 + lmod;
      const uint16_t* bp = &sBhi[cur][brow * 32 + khalf];
      v8bf b0 = *reinterpret_cast<const v8bf*>(bp);
      v8bf b1 = *reinterpret_cast<const v8bf*>(bp + 16);
      bh[t] = __builtin_shufflevector(b0, b1, 0,1,2,3,4,5,6,7,8,9,10,11,12,13,14,15);
      const uint16_t* dp = &sBlo[cur][brow * 32 + khalf];
      v8bf d0 = *reinterpret_cast<const v8bf*>(dp);
      v8bf d1 = *reinterpret_cast<const v8bf*>(dp + 16);
      bl[t] = __builtin_shufflevector(d0, d1, 0,1,2,3,4,5,6,7,8,9,10,11,12,13,14,15);
    }
#pragma unroll
    for (int m = 0; m < 2; ++m) {
#pragma unroll
      for (int n = 0; n < 2; ++n) {
        acc[m][n] = __builtin_amdgcn_wmma_f32_16x16x32_bf16(
            false, ah[m], false, bh[n], (short)0, acc[m][n], false, false);
        acc[m][n] = __builtin_amdgcn_wmma_f32_16x16x32_bf16(
            false, ah[m], false, bl[n], (short)0, acc[m][n], false, false);
        acc[m][n] = __builtin_amdgcn_wmma_f32_16x16x32_bf16(
            false, al[m], false, bh[n], (short)0, acc[m][n], false, false);
      }
    }
    __syncthreads();  // all waves done reading `cur` before TDM refills it
  }

  // C/D layout: lanes 0-15: N=lane, M=v ; lanes 16-31: N=lane-16, M=8+v
  const int nOff = lane & 15;
  const int mHi  = (lane >> 4) * 8;
#pragma unroll
  for (int m = 0; m < 2; ++m) {
#pragma unroll
    for (int n = 0; n < 2; ++n) {
      int col = colBlock + waveCol * 32 + n * 16 + nOff;
      float sqc = sq[col];
#pragma unroll
      for (int v = 0; v < 8; ++v) {
        int row = rowBlock + waveRow * 32 + m * 16 + mHi + v;
        float d2 = sq[row] + sqc - 2.0f * acc[m][n][v];
        dist[(size_t)row * NN + col] = sqrtf(fmaxf(d2, 1e-12f));
      }
    }
  }
}

// ---------- kernel 3: per-anchor mining (random positive + top-10 hard negatives) ----------
__global__ __launch_bounds__(256)
void mine_kernel(const float* __restrict__ dist, const int* __restrict__ labels,
                 float* __restrict__ d_an, int* __restrict__ negs1,
                 float* __restrict__ d_ap, int* __restrict__ valid_out) {
  __shared__ float sval[256];
  __shared__ int   sidx[256];
  __shared__ int   sneg[256];
  __shared__ int   shas[256];
  __shared__ int   chosen[K_HARD];
  __shared__ float chosenD[K_HARD];

  const int i = blockIdx.x;
  const int li = labels[i];
  const float* drow = dist + (size_t)i * NN;

  // Pass A: threefry-uniform argmax over positives, negative count, has_pos
  float bestU = -1.0f;
  int   bestJ = NN;
  int negc = 0, hasp = 0;
  for (int j = threadIdx.x; j < NN; j += 256) {
    int lj = labels[j];
    bool same = (lj == li);
    if (!same) { negc++; continue; }
    if (j == i) continue;
    hasp = 1;
    uint32_t f = (uint32_t)i * (uint32_t)NN + (uint32_t)j;
    uint32_t p = (f < HALF_CNT) ? f : (f - HALF_CNT);
    uint32_t o0, o1;
    threefry2x32(p, p + HALF_CNT, o0, o1);
    float u = jax_uniform01((f < HALF_CNT) ? o0 : o1);
    if (u > bestU || (u == bestU && j < bestJ)) { bestU = u; bestJ = j; }
  }
  sval[threadIdx.x] = bestU; sidx[threadIdx.x] = bestJ;
  sneg[threadIdx.x] = negc;  shas[threadIdx.x] = hasp;
  __syncthreads();
  for (int s = 128; s > 0; s >>= 1) {
    if ((int)threadIdx.x < s) {
      if (sval[threadIdx.x + s] > sval[threadIdx.x] ||
          (sval[threadIdx.x + s] == sval[threadIdx.x] &&
           sidx[threadIdx.x + s] < sidx[threadIdx.x])) {
        sval[threadIdx.x] = sval[threadIdx.x + s];
        sidx[threadIdx.x] = sidx[threadIdx.x + s];
      }
      sneg[threadIdx.x] += sneg[threadIdx.x + s];
      shas[threadIdx.x] |= shas[threadIdx.x + s];
    }
    __syncthreads();
  }
  if (threadIdx.x == 0) {
    int pidx = (sidx[0] >= NN) ? 0 : sidx[0];  // argmax over all -1 -> 0
    d_ap[i] = drow[pidx];
    valid_out[i] = (shas[0] && sneg[0] >= 2) ? 1 : 0;
  }
  __syncthreads();

  // Pass B: iterative 10x argmin over negatives (matches top_k tie order)
  for (int t = 0; t < K_HARD; ++t) {
    float bv = INFINITY;
    int   bj = NN;
    for (int j = threadIdx.x; j < NN; j += 256) {
      if (labels[j] == li) continue;
      bool taken = false;
      for (int q = 0; q < t; ++q)
        if (chosen[q] == j) { taken = true; break; }
      if (taken) continue;
      float d = drow[j];
      if (d < bv || (d == bv && j < bj)) { bv = d; bj = j; }
    }
    sval[threadIdx.x] = bv; sidx[threadIdx.x] = bj;
    __syncthreads();
    for (int s = 128; s > 0; s >>= 1) {
      if ((int)threadIdx.x < s) {
        if (sval[threadIdx.x + s] < sval[threadIdx.x] ||
            (sval[threadIdx.x + s] == sval[threadIdx.x] &&
             sidx[threadIdx.x + s] < sidx[threadIdx.x])) {
          sval[threadIdx.x] = sval[threadIdx.x + s];
          sidx[threadIdx.x] = sidx[threadIdx.x + s];
        }
      }
      __syncthreads();
    }
    if (threadIdx.x == 0) { chosen[t] = sidx[0]; chosenD[t] = sval[0]; }
    __syncthreads();
  }
  if (threadIdx.x < K_HARD) {
    int c = chosen[threadIdx.x];
    d_an[i * K_HARD + threadIdx.x] = chosenD[threadIdx.x];
    negs1[i * K_HARD + threadIdx.x] = (c >= NN) ? 0 : c;  // clamp; filtered by inf d_an
  }
}

// ---------- kernel 4: neg2 min-scan per (anchor, hard-neg) + term accumulation ----------
__global__ __launch_bounds__(256)
void neg2_kernel(const float* __restrict__ dist, const int* __restrict__ labels,
                 const float* __restrict__ d_an, const int* __restrict__ negs1,
                 const float* __restrict__ d_ap, const int* __restrict__ valid,
                 float* __restrict__ accum) {
  const int lane = threadIdx.x & 31;
  const int wave = threadIdx.x >> 5;
  const int pair = blockIdx.x * 8 + wave;   // pair = i*K_HARD + k
  const int i = pair / K_HARD;
  const int li = labels[i];
  const int n1 = negs1[pair];
  const float dan = d_an[pair];

  float m = INFINITY;
  if (finitef(dan) && valid[i]) {
    const float* row = dist + (size_t)n1 * NN;
    for (int j = lane; j < NN; j += 32) {
      if (labels[j] != li && j != n1) m = fminf(m, row[j]);
    }
  }
  for (int off = 16; off > 0; off >>= 1) m = fminf(m, __shfl_xor(m, off, 32));

  if (lane == 0) {
    bool v = valid[i] && finitef(dan) && finitef(m);
    if (v) {
      float dap = d_ap[i];
      float term = fmaxf(dap - dan + MARGIN1, 0.f) + fmaxf(dap - m + MARGIN2, 0.f);
      atomicAdd(&accum[0], term);
      atomicAdd(&accum[1], 1.0f);
    }
  }
}

// ---------- kernel 5: center loss ----------
__global__ __launch_bounds__(256)
void center_kernel(const float* __restrict__ X, const int* __restrict__ labels,
                   const float* __restrict__ centers, float* __restrict__ accum) {
  __shared__ float red[256];
  float acc = 0.f;
  const size_t total = (size_t)NN * DD;
  for (size_t idx = (size_t)blockIdx.x * 256 + threadIdx.x; idx < total;
       idx += (size_t)gridDim.x * 256) {
    int i = (int)(idx / DD);
    int d = (int)(idx % DD);
    float diff = X[idx] - centers[(size_t)labels[i] * DD + d];
    acc += diff * diff;
  }
  red[threadIdx.x] = acc;
  __syncthreads();
  for (int s = 128; s > 0; s >>= 1) {
    if ((int)threadIdx.x < s) red[threadIdx.x] += red[threadIdx.x + s];
    __syncthreads();
  }
  if (threadIdx.x == 0) atomicAdd(&accum[2], red[0]);
}

// ---------- kernel 6: finalize ----------
__global__ void finalize_kernel(const float* __restrict__ accum, float* __restrict__ out) {
  float total = accum[0], cnt = accum[1], center = accum[2];
  float quad = (cnt > 0.f) ? (total / cnt) : 0.f;
  out[0] = quad + CENTER_W * (center / (float)NN);
}

// ---------- launch ----------
extern "C" void kernel_launch(void* const* d_in, const int* in_sizes, int n_in,
                              void* d_out, int out_size, void* d_ws, size_t ws_size,
                              hipStream_t stream) {
  const float* X       = (const float*)d_in[0];
  const int*   labels  = (const int*)d_in[1];
  const float* centers = (const float*)d_in[2];
  float* out = (float*)d_out;

  char* ws = (char*)d_ws;
  const size_t szHalfBf = (size_t)NN * DD * sizeof(uint16_t);  // 8 MB
  const size_t szDist   = (size_t)NN * NN * sizeof(float);     // 16 MB
  uint16_t* Xhi   = (uint16_t*)(ws);
  uint16_t* Xlo   = (uint16_t*)(ws + szHalfBf);
  float*    dist  = (float*)(ws + 2 * szHalfBf);
  float*    sq    = (float*)(ws + 2 * szHalfBf + szDist);
  float*    d_an  = (float*)((char*)sq + NN * sizeof(float));
  int*      negs1 = (int*)((char*)d_an + NN * K_HARD * sizeof(float));
  float*    d_ap  = (float*)((char*)negs1 + NN * K_HARD * sizeof(int));
  int*      valid = (int*)((char*)d_ap + NN * sizeof(float));
  float*    accum = (float*)((char*)valid + NN * sizeof(int));

  init_kernel<<<1, 32, 0, stream>>>(accum);
  prep_kernel<<<NN, 256, 0, stream>>>(X, Xhi, Xlo, sq);
  gram_dist_kernel<<<dim3(NN / 64, NN / 128), 256, 0, stream>>>(Xhi, Xlo, sq, dist);
  mine_kernel<<<NN, 256, 0, stream>>>(dist, labels, d_an, negs1, d_ap, valid);
  neg2_kernel<<<(NN * K_HARD) / 8, 256, 0, stream>>>(dist, labels, d_an, negs1, d_ap, valid, accum);
  center_kernel<<<1024, 256, 0, stream>>>(X, labels, centers, accum);
  finalize_kernel<<<1, 1, 0, stream>>>(accum, out);
}
